// Str2Str_3307124817996
// MI455X (gfx1250) — compile-verified, matrix operand order checked by reading, not run on hardware
//
#include <hip/hip_runtime.h>
#include <hip/hip_bf16.h>

// ---------------------------------------------------------------------------
// CDNA5 (gfx1250) wave32 WMMA helpers: v_wmma_f32_16x16x32_bf16
// Fragment layouts per CDNA5 ISA 7.12.2 (16-bit A 16x32, C/D 32-bit 16x16).
// ---------------------------------------------------------------------------
typedef __attribute__((ext_vector_type(16))) __bf16          v16bf;
typedef __attribute__((ext_vector_type(16))) unsigned short  v16us;
typedef __attribute__((ext_vector_type(8)))  float           v8f;

static __device__ inline unsigned short f2bf(float f) {
    unsigned int u = __float_as_uint(f);
    u += 0x7fffu + ((u >> 16) & 1u);          // round-to-nearest-even
    return (unsigned short)(u >> 16);
}

static __device__ inline v8f zero8() {
    v8f z;
#pragma unroll
    for (int i = 0; i < 8; ++i) z[i] = 0.f;
    return z;
}

// A fragment (16xK tile, bf16 stored as ushort in LDS, row-major ld=ldA), k-chunk k0..k0+31
static __device__ inline v16bf load_a_frag(const unsigned short* A, int ldA, int k0) {
    int lane = threadIdx.x & 31;
    int m  = lane & 15;
    int kb = k0 + ((lane >> 4) << 3);
    v16us u;
#pragma unroll
    for (int h = 0; h < 8; ++h) {
        u[h]     = A[m * ldA + kb + h];
        u[h + 8] = A[m * ldA + kb + 16 + h];
    }
    return __builtin_bit_cast(v16bf, u);
}

// B fragment from global f32 weights (KxN row-major, ld=ldb), zero-padded past kmax/nmax
static __device__ inline v16bf load_b_frag(const float* W, int ldb, int nmax,
                                           int n0, int k0, int kmax) {
    int lane = threadIdx.x & 31;
    int n  = n0 + (lane & 15);
    int kb = k0 + ((lane >> 4) << 3);
    v16us u;
#pragma unroll
    for (int h = 0; h < 16; ++h) {
        int K = kb + ((h < 8) ? h : (h + 8));
        float w = 0.f;
        if (K < kmax && n < nmax) w = W[(size_t)K * ldb + n];
        u[h] = f2bf(w);
    }
    return __builtin_bit_cast(v16bf, u);
}

// Combined B for edge: rows 0..127 = ee1_w(128x32), 128..192 = ee2_w(65x32), rest zero
static __device__ inline v16bf load_b_comb(const float* w1, const float* w2, int n0, int k0) {
    int lane = threadIdx.x & 31;
    int n  = n0 + (lane & 15);
    int kb = k0 + ((lane >> 4) << 3);
    v16us u;
#pragma unroll
    for (int h = 0; h < 16; ++h) {
        int K = kb + ((h < 8) ? h : (h + 8));
        float w = 0.f;
        if (K < 128)       w = w1[K * 32 + n];
        else if (K < 193)  w = w2[(K - 128) * 32 + n];
        u[h] = f2bf(w);
    }
    return __builtin_bit_cast(v16bf, u);
}

static __device__ inline v8f wmma_bf16(v16bf a, v16bf b, v8f c) {
    return __builtin_amdgcn_wmma_f32_16x16x32_bf16(false, a, false, b, (short)0, c, false, false);
}

// ---------------------------------------------------------------------------
// Kernel A: node features + normalized state.  node = ln-resid MLP, 32-dim.
// One 64-thread block per residue.
// ---------------------------------------------------------------------------
__global__ void node_kernel(const float* __restrict__ msa, const float* __restrict__ state,
                            const float* nmg, const float* nmb,
                            const float* nsg, const float* nsb,
                            const float* en1w, const float* en1b,
                            const float* en2w, const float* en2b,
                            const float* fng, const float* fnb,
                            const float* fw1, const float* fb1,
                            const float* fw2, const float* fb2,
                            const float* nng, const float* nnb,
                            float* __restrict__ node_out, float* __restrict__ stn_out) {
    int l = blockIdx.x, tid = threadIdx.x;
    __shared__ float xs[256];
    __shared__ float red[64];
    __shared__ float sm[2];
    __shared__ float stn[16];
    __shared__ float nodep[32];
    __shared__ float nrm[32];
    __shared__ float h1[64];

    const float* xrow = msa + (size_t)l * 256;
    float ls = 0.f, lss = 0.f;
    for (int t = tid; t < 256; t += 64) { float v = xrow[t]; xs[t] = v; ls += v; lss += v * v; }
    red[tid] = ls; __syncthreads();
    for (int s = 32; s > 0; s >>= 1) { if (tid < s) red[tid] += red[tid + s]; __syncthreads(); }
    float mean = red[0] / 256.f; __syncthreads();
    red[tid] = lss; __syncthreads();
    for (int s = 32; s > 0; s >>= 1) { if (tid < s) red[tid] += red[tid + s]; __syncthreads(); }
    if (tid == 0) { float var = red[0] / 256.f - mean * mean; sm[0] = mean; sm[1] = rsqrtf(var + 1e-5f); }
    __syncthreads();
    float m = sm[0], inv = sm[1];
    for (int t = tid; t < 256; t += 64) xs[t] = (xs[t] - m) * inv * nmg[t] + nmb[t];
    if (tid == 0) {
        const float* st = state + (size_t)l * 16;
        float s = 0.f, ss = 0.f;
        for (int i = 0; i < 16; ++i) { s += st[i]; ss += st[i] * st[i]; }
        float mm = s / 16.f, iv = rsqrtf(ss / 16.f - mm * mm + 1e-5f);
        for (int i = 0; i < 16; ++i) {
            float v = (st[i] - mm) * iv * nsg[i] + nsb[i];
            stn[i] = v; stn_out[(size_t)l * 16 + i] = v;
        }
    }
    __syncthreads();
    if (tid < 32) {
        float s = en1b[tid] + en2b[tid];
        for (int i = 0; i < 256; ++i) s += xs[i] * en1w[i * 32 + tid];
        for (int i = 0; i < 16; ++i)  s += stn[i] * en2w[i * 32 + tid];
        nodep[tid] = s;
    }
    __syncthreads();
    if (tid == 0) {
        float s = 0.f, ss = 0.f;
        for (int i = 0; i < 32; ++i) { s += nodep[i]; ss += nodep[i] * nodep[i]; }
        float mm = s / 32.f; sm[0] = mm; sm[1] = rsqrtf(ss / 32.f - mm * mm + 1e-5f);
    }
    __syncthreads();
    if (tid < 32) nrm[tid] = (nodep[tid] - sm[0]) * sm[1] * fng[tid] + fnb[tid];
    __syncthreads();
    { float s = fb1[tid]; for (int j = 0; j < 32; ++j) s += nrm[j] * fw1[j * 64 + tid]; h1[tid] = fmaxf(s, 0.f); }
    __syncthreads();
    if (tid < 32) {
        float s = fb2[tid];
        for (int k = 0; k < 64; ++k) s += h1[k] * fw2[k * 32 + tid];
        nodep[tid] += s;
    }
    __syncthreads();
    if (tid == 0) {
        float s = 0.f, ss = 0.f;
        for (int i = 0; i < 32; ++i) { s += nodep[i]; ss += nodep[i] * nodep[i]; }
        float mm = s / 32.f; sm[0] = mm; sm[1] = rsqrtf(ss / 32.f - mm * mm + 1e-5f);
    }
    __syncthreads();
    if (tid < 32) node_out[(size_t)l * 32 + tid] = (nodep[tid] - sm[0]) * sm[1] * nng[tid] + nnb[tid];
}

// ---------------------------------------------------------------------------
// Kernel B: top-64 neighbors by Dg = D + 999.9*eye + 1e-5*|seqsep| (ascending),
// matching jax.lax.top_k(-Dg) ordering + tie-breaking (lower index first).
// ---------------------------------------------------------------------------
__global__ void topk_kernel(const float* __restrict__ xyz, const int* __restrict__ idx,
                            int* __restrict__ nbr) {
    int l = blockIdx.x, tid = threadIdx.x;
    __shared__ float dg[768];
    __shared__ float rv[256];
    __shared__ int   ri[256];
    float c0 = xyz[(size_t)l * 9 + 3], c1 = xyz[(size_t)l * 9 + 4], c2 = xyz[(size_t)l * 9 + 5];
    int il = idx[l];
    for (int j = tid; j < 768; j += 256) {
        float d0 = xyz[(size_t)j * 9 + 3] - c0;
        float d1 = xyz[(size_t)j * 9 + 4] - c1;
        float d2 = xyz[(size_t)j * 9 + 5] - c2;
        float D = sqrtf(d0 * d0 + d1 * d1 + d2 * d2 + 1e-8f);
        dg[j] = D + ((j == l) ? 999.9f : 0.f) + fabsf((float)(idx[j] - il)) * 1e-5f;
    }
    __syncthreads();
    for (int it = 0; it < 64; ++it) {
        float bv = 3.0e38f; int bi = 0x7fffffff;
        for (int j = tid; j < 768; j += 256) {
            float v = dg[j];
            if (v < bv) { bv = v; bi = j; }
        }
        rv[tid] = bv; ri[tid] = bi; __syncthreads();
        for (int s = 128; s > 0; s >>= 1) {
            if (tid < s) {
                float ov = rv[tid + s]; int oi = ri[tid + s];
                if (ov < rv[tid] || (ov == rv[tid] && oi < ri[tid])) { rv[tid] = ov; ri[tid] = oi; }
            }
            __syncthreads();
        }
        if (tid == 0) { nbr[(size_t)l * 64 + it] = ri[0]; dg[ri[0]] = 3.0e38f; }
        __syncthreads();
    }
}

// ---------------------------------------------------------------------------
// Kernel C: edge features at gathered (l, nbr) rows only (768*64 rows instead
// of 768^2).  One wave (32 lanes) per 16-row tile; WMMA bf16 GEMMs:
//   [pairLN | rbf] (16x224) @ [ee1_w;ee2_w] (224x32)   -> edge_pre
//   ffe: LN -> @w1(32x64) relu -> @w2(64x32) residual -> LN -> efeat
// ---------------------------------------------------------------------------
__global__ void edge_kernel(const float* __restrict__ pair, const float* __restrict__ xyz,
                            const int* __restrict__ idx, const int* __restrict__ nbr,
                            const float* npg, const float* npb,
                            const float* ee1w, const float* ee1b,
                            const float* ee2w, const float* ee2b,
                            const float* feng, const float* fenb,
                            const float* few1, const float* feb1,
                            const float* few2, const float* feb2,
                            const float* negg, const float* negb,
                            float* __restrict__ efeat) {
    int l    = blockIdx.x >> 2;
    int kt   = (blockIdx.x & 3) << 4;
    int lane = threadIdx.x;
    __shared__ float          raw[16 * 128];
    __shared__ unsigned short Abf[16 * 224];
    __shared__ float          epre[16 * 32];
    __shared__ float          eout[16 * 32];
    __shared__ int            jn[16];
    __shared__ float          Drow[16], nbv[16];
    __shared__ float          rm[16], rinv[16];

    if (lane < 16) {
        int j = nbr[(size_t)l * 64 + kt + lane];
        jn[lane] = j;
        __builtin_prefetch(pair + ((size_t)l * 768 + j) * 128, 0, 0);  // global_prefetch_b8
        float d0 = xyz[(size_t)j * 9 + 3] - xyz[(size_t)l * 9 + 3];
        float d1 = xyz[(size_t)j * 9 + 4] - xyz[(size_t)l * 9 + 4];
        float d2 = xyz[(size_t)j * 9 + 5] - xyz[(size_t)l * 9 + 5];
        Drow[lane] = sqrtf(d0 * d0 + d1 * d1 + d2 * d2 + 1e-8f);
        float sep = (float)(idx[j] - idx[l]);
        float sg = (sep > 0.f) ? 1.f : ((sep < 0.f) ? -1.f : 0.f);
        nbv[lane] = sg * logf(fabsf(sep) + 1.f);
    }
    __syncthreads();
    for (int t = lane; t < 16 * 128; t += 32) {
        int r = t >> 7, c = t & 127;
        raw[t] = pair[((size_t)l * 768 + jn[r]) * 128 + c];
    }
    __syncthreads();
    if (lane < 16) {  // layernorm pair row -> bf16 A columns 0..127
        const float* x = raw + lane * 128;
        float s = 0.f, ss = 0.f;
        for (int c = 0; c < 128; ++c) { float v = x[c]; s += v; ss += v * v; }
        float m = s * (1.f / 128.f), iv = rsqrtf(ss * (1.f / 128.f) - m * m + 1e-5f);
        unsigned short* a = Abf + lane * 224;
        for (int c = 0; c < 128; ++c) a[c] = f2bf((x[c] - m) * iv * npg[c] + npb[c]);
    }
    for (int t = lane; t < 16 * 96; t += 32) {  // rbf + nbrf -> A columns 128..223 (zero-padded)
        int r = t / 96, f = t - r * 96;
        float v;
        if (f < 64) { float mu = 2.f + (20.f / 63.f) * (float)f; float z = (Drow[r] - mu) * 3.2f; v = expf(-z * z); }
        else if (f == 64) v = nbv[r];
        else v = 0.f;
        Abf[r * 224 + 128 + f] = f2bf(v);
    }
    __syncthreads();

    // GEMM1: (16x224) @ (224x32)
#pragma unroll
    for (int nt = 0; nt < 2; ++nt) {
        v8f acc = zero8();
        for (int kc = 0; kc < 7; ++kc) {
            v16bf a = load_a_frag(Abf, 224, kc * 32);
            v16bf b = load_b_comb(ee1w, ee2w, nt * 16, kc * 32);
            acc = wmma_bf16(a, b, acc);
        }
        int n = nt * 16 + (lane & 15);
        int mb = (lane >> 4) << 3;
#pragma unroll
        for (int v = 0; v < 8; ++v) epre[(mb + v) * 32 + n] = acc[v] + ee1b[n] + ee2b[n];
    }
    __syncthreads();
    if (lane < 16) {  // ffe LN -> bf16 (ld 32)
        const float* x = epre + lane * 32;
        float s = 0.f, ss = 0.f;
        for (int c = 0; c < 32; ++c) { s += x[c]; ss += x[c] * x[c]; }
        float m = s / 32.f, iv = rsqrtf(ss / 32.f - m * m + 1e-5f);
        for (int c = 0; c < 32; ++c) Abf[lane * 32 + c] = f2bf((x[c] - m) * iv * feng[c] + fenb[c]);
    }
    __syncthreads();
    unsigned short* H = Abf + 16 * 32;  // hidden 16x64 bf16
#pragma unroll
    for (int nt = 0; nt < 4; ++nt) {
        v16bf a = load_a_frag(Abf, 32, 0);
        v16bf b = load_b_frag(few1, 64, 64, nt * 16, 0, 32);
        v8f acc = wmma_bf16(a, b, zero8());
        int n = nt * 16 + (lane & 15);
        int mb = (lane >> 4) << 3;
#pragma unroll
        for (int v = 0; v < 8; ++v) H[(mb + v) * 64 + n] = f2bf(fmaxf(acc[v] + feb1[n], 0.f));
    }
    __syncthreads();
#pragma unroll
    for (int nt = 0; nt < 2; ++nt) {
        v8f acc = zero8();
        for (int kc = 0; kc < 2; ++kc) {
            v16bf a = load_a_frag(H, 64, kc * 32);
            v16bf b = load_b_frag(few2, 32, 32, nt * 16, kc * 32, 64);
            acc = wmma_bf16(a, b, acc);
        }
        int n = nt * 16 + (lane & 15);
        int mb = (lane >> 4) << 3;
#pragma unroll
        for (int v = 0; v < 8; ++v) eout[(mb + v) * 32 + n] = epre[(mb + v) * 32 + n] + acc[v] + feb2[n];
    }
    __syncthreads();
    if (lane < 16) {
        const float* x = eout + lane * 32;
        float s = 0.f, ss = 0.f;
        for (int c = 0; c < 32; ++c) { s += x[c]; ss += x[c] * x[c]; }
        float m = s / 32.f; rm[lane] = m; rinv[lane] = rsqrtf(ss / 32.f - m * m + 1e-5f);
    }
    __syncthreads();
    for (int t = lane; t < 512; t += 32) {
        int r = t >> 5, c = t & 31;
        efeat[((size_t)l * 64 + kt + r) * 32 + c] = (eout[t] - rm[r]) * rinv[r] * negg[c] + negb[c];
    }
}

// ---------------------------------------------------------------------------
// Kernel D: SE3-ish head: softmax attention over 64 neighbors, shift0/shift1,
// state update, translation + quaternion rotation outputs.  64 threads/residue.
// ---------------------------------------------------------------------------
__global__ void se3_kernel(const float* __restrict__ xyz, const float* __restrict__ Rin,
                           const float* __restrict__ Tin, const int* __restrict__ nbr,
                           const float* __restrict__ node, const float* __restrict__ stn,
                           const float* __restrict__ efeat,
                           const float* watt, const float* w0, const float* b0,
                           const float* wvec, const float* wmix,
                           float* __restrict__ Rs_out, float* __restrict__ T_out,
                           float* __restrict__ st_out) {
    int l = blockIdx.x, tid = threadIdx.x;
    __shared__ float logit[64], att[64], red[64];
    __shared__ float bco[64 * 2], rel[64 * 3];
    __shared__ float nodel[32], msg0v[32];
    __shared__ int nb[64];
    __shared__ float cal[3];
    __shared__ float sh1[6];

    if (tid == 0) { cal[0] = xyz[(size_t)l * 9 + 3]; cal[1] = xyz[(size_t)l * 9 + 4]; cal[2] = xyz[(size_t)l * 9 + 5]; }
    nb[tid] = nbr[(size_t)l * 64 + tid];
    __syncthreads();
    {
        const float* e = efeat + ((size_t)l * 64 + tid) * 32;
        float s = 0.f, c0 = 0.f, c1 = 0.f;
        for (int c = 0; c < 32; ++c) { float v = e[c]; s += v * watt[c]; c0 += v * wvec[c * 2]; c1 += v * wvec[c * 2 + 1]; }
        logit[tid] = s; bco[tid * 2] = c0; bco[tid * 2 + 1] = c1;
        int j = nb[tid];
        rel[tid * 3 + 0] = xyz[(size_t)j * 9 + 3] - cal[0];
        rel[tid * 3 + 1] = xyz[(size_t)j * 9 + 4] - cal[1];
        rel[tid * 3 + 2] = xyz[(size_t)j * 9 + 5] - cal[2];
    }
    __syncthreads();
    red[tid] = logit[tid]; __syncthreads();
    for (int s = 32; s > 0; s >>= 1) { if (tid < s) red[tid] = fmaxf(red[tid], red[tid + s]); __syncthreads(); }
    float mx = red[0]; __syncthreads();
    float ev = expf(logit[tid] - mx);
    red[tid] = ev; att[tid] = ev; __syncthreads();
    for (int s = 32; s > 0; s >>= 1) { if (tid < s) red[tid] += red[tid + s]; __syncthreads(); }
    float denom = red[0]; __syncthreads();
    att[tid] /= denom;
    if (tid < 32) nodel[tid] = node[(size_t)l * 32 + tid];
    __syncthreads();
    if (tid < 32) {
        float s = 0.f;
        for (int k = 0; k < 64; ++k) s += att[k] * node[(size_t)nb[k] * 32 + tid];
        msg0v[tid] = s;
    }
    __syncthreads();
    if (tid < 16) {
        float s = b0[tid];
        for (int c = 0; c < 32; ++c) s += (nodel[c] + msg0v[c]) * w0[c * 16 + tid];
        st_out[(size_t)l * 16 + tid] = stn[(size_t)l * 16 + tid] + s;
    }
    if (tid < 6) {
        int o = tid / 3, d = tid - 3 * o;
        float s = 0.f;
        for (int k = 0; k < 64; ++k) s += bco[k * 2 + o] * rel[k * 3 + d];
        float l10 = xyz[(size_t)l * 9 + 0 + d] - cal[d];   // xyz[:, :, 0]
        float l11 = xyz[(size_t)l * 9 + 6 + d] - cal[d];   // xyz[:, :, 2]
        sh1[tid] = wmix[o * 2 + 0] * l10 + wmix[o * 2 + 1] * l11 + 0.1f * s;
    }
    __syncthreads();
    if (tid == 0) {
        for (int d = 0; d < 3; ++d) T_out[(size_t)l * 3 + d] = sh1[d] * 10.f + Tin[(size_t)l * 3 + d];
        float qx = sh1[3], qy = sh1[4], qz = sh1[5];
        float inv = rsqrtf(1.f + qx * qx + qy * qy + qz * qz);
        float w = inv, x = qx * inv, y = qy * inv, z = qz * inv;
        float Rq[9] = {
            1.f - 2.f * (y * y + z * z), 2.f * (x * y - w * z),       2.f * (x * z + w * y),
            2.f * (x * y + w * z),       1.f - 2.f * (x * x + z * z), 2.f * (y * z - w * x),
            2.f * (x * z - w * y),       2.f * (y * z + w * x),       1.f - 2.f * (x * x + y * y)
        };
        const float* R = Rin + (size_t)l * 9;
        for (int i = 0; i < 3; ++i)
            for (int k = 0; k < 3; ++k) {
                float s = 0.f;
                for (int j = 0; j < 3; ++j) s += Rq[i * 3 + j] * R[j * 3 + k];
                Rs_out[(size_t)l * 9 + i * 3 + k] = s;
            }
    }
}

// ---------------------------------------------------------------------------
// Kernel E: side-chain predictor MLP with WMMA bf16 (16 rows per wave):
// si = ln(s0)@W(256x128) + ln(st)@W(16x128); two 128x128 relu-resid blocks;
// out = relu(si)@Wout(128x20).
// ---------------------------------------------------------------------------
__global__ void scpred_kernel(const float* __restrict__ msa, const float* __restrict__ stfull,
                              const float* n0g, const float* n0b,
                              const float* nig, const float* nib,
                              const float* s0w, const float* s0b,
                              const float* siw, const float* sib,
                              const float* w1, const float* b1, const float* w2, const float* b2,
                              const float* w3, const float* b3, const float* w4, const float* b4,
                              const float* wout, const float* bout,
                              float* __restrict__ alpha) {
    int l0 = blockIdx.x << 4;
    int lane = threadIdx.x;
    __shared__ unsigned short A0[16 * 256];
    __shared__ unsigned short A1[16 * 128];
    __shared__ unsigned short A2[16 * 32];
    __shared__ float          si[16 * 128];

    if (lane < 16) {
        const float* x = msa + (size_t)(l0 + lane) * 256;
        float s = 0.f, ss = 0.f;
        for (int c = 0; c < 256; ++c) { float v = x[c]; s += v; ss += v * v; }
        float m = s / 256.f, iv = rsqrtf(ss / 256.f - m * m + 1e-5f);
        unsigned short* a = A0 + lane * 256;
        for (int c = 0; c < 256; ++c) a[c] = f2bf((x[c] - m) * iv * n0g[c] + n0b[c]);
        const float* st = stfull + (size_t)(l0 + lane) * 16;
        float s2 = 0.f, ss2 = 0.f;
        for (int c = 0; c < 16; ++c) { s2 += st[c]; ss2 += st[c] * st[c]; }
        float m2 = s2 / 16.f, iv2 = rsqrtf(ss2 / 16.f - m2 * m2 + 1e-5f);
        unsigned short* a2 = A2 + lane * 32;
        for (int c = 0; c < 16; ++c) a2[c] = f2bf((st[c] - m2) * iv2 * nig[c] + nib[c]);
        for (int c = 16; c < 32; ++c) a2[c] = 0;
    }
    __syncthreads();
    for (int nt = 0; nt < 8; ++nt) {
        v8f acc = zero8();
        for (int kc = 0; kc < 8; ++kc)
            acc = wmma_bf16(load_a_frag(A0, 256, kc * 32),
                            load_b_frag(s0w, 128, 128, nt * 16, kc * 32, 256), acc);
        acc = wmma_bf16(load_a_frag(A2, 32, 0),
                        load_b_frag(siw, 128, 128, nt * 16, 0, 16), acc);
        int n = nt * 16 + (lane & 15);
        int mb = (lane >> 4) << 3;
#pragma unroll
        for (int v = 0; v < 8; ++v) si[(mb + v) * 128 + n] = acc[v] + s0b[n] + sib[n];
    }
    __syncthreads();
    for (int blk = 0; blk < 2; ++blk) {
        const float* wa = blk ? w3 : w1; const float* ba = blk ? b3 : b1;
        const float* wb = blk ? w4 : w2; const float* bb = blk ? b4 : b2;
        for (int t = lane; t < 16 * 128; t += 32) A1[t] = f2bf(fmaxf(si[t], 0.f));
        __syncthreads();
        for (int nt = 0; nt < 8; ++nt) {
            v8f acc = zero8();
            for (int kc = 0; kc < 4; ++kc)
                acc = wmma_bf16(load_a_frag(A1, 128, kc * 32),
                                load_b_frag(wa, 128, 128, nt * 16, kc * 32, 128), acc);
            int n = nt * 16 + (lane & 15);
            int mb = (lane >> 4) << 3;
#pragma unroll
            for (int v = 0; v < 8; ++v) A0[(mb + v) * 128 + n] = f2bf(fmaxf(acc[v] + ba[n], 0.f));
        }
        __syncthreads();
        for (int nt = 0; nt < 8; ++nt) {
            v8f acc = zero8();
            for (int kc = 0; kc < 4; ++kc)
                acc = wmma_bf16(load_a_frag(A0, 128, kc * 32),
                                load_b_frag(wb, 128, 128, nt * 16, kc * 32, 128), acc);
            int n = nt * 16 + (lane & 15);
            int mb = (lane >> 4) << 3;
#pragma unroll
            for (int v = 0; v < 8; ++v) si[(mb + v) * 128 + n] += acc[v] + bb[n];
        }
        __syncthreads();
    }
    for (int t = lane; t < 16 * 128; t += 32) A1[t] = f2bf(fmaxf(si[t], 0.f));
    __syncthreads();
    for (int nt = 0; nt < 2; ++nt) {
        v8f acc = zero8();
        for (int kc = 0; kc < 4; ++kc)
            acc = wmma_bf16(load_a_frag(A1, 128, kc * 32),
                            load_b_frag(wout, 20, 20, nt * 16, kc * 32, 128), acc);
        int n = nt * 16 + (lane & 15);
        int mb = (lane >> 4) << 3;
#pragma unroll
        for (int v = 0; v < 8; ++v)
            if (n < 20) alpha[(size_t)(l0 + mb + v) * 20 + n] = acc[v] + bout[n];
    }
}

// ---------------------------------------------------------------------------
// Host launcher
// ---------------------------------------------------------------------------
extern "C" void kernel_launch(void* const* d_in, const int* in_sizes, int n_in,
                              void* d_out, int out_size, void* d_ws, size_t ws_size,
                              hipStream_t stream) {
    (void)in_sizes; (void)n_in; (void)out_size; (void)ws_size;
    const float* msa   = (const float*)d_in[0];
    const float* pair  = (const float*)d_in[1];
    const float* Rin   = (const float*)d_in[2];
    const float* Tin   = (const float*)d_in[3];
    const float* xyz   = (const float*)d_in[4];
    const float* state = (const float*)d_in[5];
    const int*   idx   = (const int*)d_in[6];
    // d_in[7] = top_k (fixed 64)

    // params: jax pytree (dict) flattening = sorted key order, leaves at d_in[8..60]
    const float* P[53];
    for (int i = 0; i < 53; ++i) P[i] = (const float*)d_in[8 + i];
    const float *ee1_b = P[0], *ee1_w = P[1], *ee2_b = P[2], *ee2_w = P[3];
    const float *en1_b = P[4], *en1_w = P[5], *en2_b = P[6], *en2_w = P[7];
    const float *ffe_b1 = P[8], *ffe_b2 = P[9], *ffe_nb = P[10], *ffe_ng = P[11];
    const float *ffe_w1 = P[12], *ffe_w2 = P[13];
    const float *ffn_b1 = P[14], *ffn_b2 = P[15], *ffn_nb = P[16], *ffn_ng = P[17];
    const float *ffn_w1 = P[18], *ffn_w2 = P[19];
    const float *norm_edge_b = P[20], *norm_edge_g = P[21];
    const float *norm_msa_b = P[22], *norm_msa_g = P[23];
    const float *norm_node_b = P[24], *norm_node_g = P[25];
    const float *norm_pair_b = P[26], *norm_pair_g = P[27];
    const float *norm_state_b = P[28], *norm_state_g = P[29];
    const float *sc_b1 = P[30], *sc_b2 = P[31], *sc_b3 = P[32], *sc_b4 = P[33], *sc_bout = P[34];
    const float *sc_n0_b = P[35], *sc_n0_g = P[36], *sc_ni_b = P[37], *sc_ni_g = P[38];
    const float *sc_s0_b = P[39], *sc_s0_w = P[40], *sc_si_b = P[41], *sc_si_w = P[42];
    const float *sc_w1 = P[43], *sc_w2 = P[44], *sc_w3 = P[45], *sc_w4 = P[46], *sc_wout = P[47];
    const float *se3_b0 = P[48], *se3_w0 = P[49], *se3_watt = P[50];
    const float *se3_wmix = P[51], *se3_wvec = P[52];

    float* out    = (float*)d_out;
    float* Rs_out = out;             // 768*9
    float* T_out  = out + 6912;      // 768*3
    float* st_out = out + 9216;      // 768*16
    float* alpha  = out + 21504;     // 768*20

    char* ws = (char*)d_ws;
    float* ws_node  = (float*)ws; ws += (size_t)768 * 32 * sizeof(float);
    float* ws_stn   = (float*)ws; ws += (size_t)768 * 16 * sizeof(float);
    int*   ws_nbr   = (int*)ws;   ws += (size_t)768 * 64 * sizeof(int);
    float* ws_efeat = (float*)ws; ws += (size_t)768 * 64 * 32 * sizeof(float);

    node_kernel<<<768, 64, 0, stream>>>(msa, state, norm_msa_g, norm_msa_b,
        norm_state_g, norm_state_b, en1_w, en1_b, en2_w, en2_b,
        ffn_ng, ffn_nb, ffn_w1, ffn_b1, ffn_w2, ffn_b2,
        norm_node_g, norm_node_b, ws_node, ws_stn);
    topk_kernel<<<768, 256, 0, stream>>>(xyz, idx, ws_nbr);
    edge_kernel<<<3072, 32, 0, stream>>>(pair, xyz, idx, ws_nbr,
        norm_pair_g, norm_pair_b, ee1_w, ee1_b, ee2_w, ee2_b,
        ffe_ng, ffe_nb, ffe_w1, ffe_b1, ffe_w2, ffe_b2,
        norm_edge_g, norm_edge_b, ws_efeat);
    se3_kernel<<<768, 64, 0, stream>>>(xyz, Rin, Tin, ws_nbr, ws_node, ws_stn, ws_efeat,
        se3_watt, se3_w0, se3_b0, se3_wvec, se3_wmix, Rs_out, T_out, st_out);
    scpred_kernel<<<48, 32, 0, stream>>>(msa, st_out,
        sc_n0_g, sc_n0_b, sc_ni_g, sc_ni_b, sc_s0_w, sc_s0_b, sc_si_w, sc_si_b,
        sc_w1, sc_b1, sc_w2, sc_b2, sc_w3, sc_b3, sc_w4, sc_b4, sc_wout, sc_bout, alpha);
}
